// VOS_16690242912330
// MI455X (gfx1250) — compile-verified, matrix operand order checked by reading, not run on hardware
//
#include <hip/hip_runtime.h>
#include <hip/hip_bf16.h>
#include <stdint.h>

// ---------------------------------------------------------------------------
// Types for WMMA (gfx1250, wave32) + TDM descriptor vectors
// ---------------------------------------------------------------------------
typedef _Float16 v16h __attribute__((ext_vector_type(16)));
typedef _Float16 v8h  __attribute__((ext_vector_type(8)));
typedef _Float16 v4h  __attribute__((ext_vector_type(4)));
typedef float    v8f  __attribute__((ext_vector_type(8)));

typedef unsigned int u32x4 __attribute__((ext_vector_type(4)));
typedef int          i32x4 __attribute__((ext_vector_type(4)));
typedef int          i32x8 __attribute__((ext_vector_type(8)));

#if defined(__has_builtin)
#if __has_builtin(__builtin_amdgcn_tensor_load_to_lds) && \
    __has_builtin(__builtin_amdgcn_s_wait_tensorcnt)
#define ATHENA_TDM 1
#endif
#endif

#define TM 64
#define TN 128
#define TK 32
#define LDSK 40   // padded K stride (halves): 80B row pitch keeps 16B alignment

// ---------------------------------------------------------------------------
// Generic GEMM: C[M,N] = op(A) * B, f32 in/out, f16 WMMA accumulate-to-f32.
//   ATRANS=1: A is K x M (A[k*lda+m])   (similarity: key tensors are C x HW)
//   ATRANS=0: A is M x K (A[m*lda+k])   (conv weights)
//   CONV3=1 : B gathered as implicit 3x3 conv, K = inC*9, B is inC x Hd x Wd
//   EPI: 1 = bias + relu, 2 = affine 0.5*x + 0.5  (cosine sim -> [0,1])
// Block: 256 threads = 8 waves (2x4), each wave owns a 32x32 output tile
// (4 x v_wmma_f32_16x16x32_f16 per K-step).
// ---------------------------------------------------------------------------
template <int ATRANS, int CONV3, int EPI>
__global__ __launch_bounds__(256)
void gemm_wmma_f16(const float* __restrict__ A, long long aBatch, int lda,
                   const float* __restrict__ Bm, long long bBatch,
                   float* __restrict__ C, long long cBatch, int ldc,
                   int M, int N, int K,
                   const float* __restrict__ bias,
                   int Hd, int Wd)
{
    __shared__ __align__(16) _Float16 As[TM * LDSK];
    __shared__ __align__(16) _Float16 Bs[TN * LDSK];

    const int tid  = threadIdx.x;
    const int lane = tid & 31;
    const int wave = tid >> 5;
    const int wm   = wave & 1;   // 0..1 -> M
    const int wn   = wave >> 1;  // 0..3 -> N
    const int bz   = blockIdx.z;
    const int n0   = blockIdx.x * TN;
    const int m0   = blockIdx.y * TM;

    const float* Ab = A  + (size_t)bz * aBatch;
    const float* Bb = Bm + (size_t)bz * bBatch;
    float*       Cb = C  + (size_t)bz * cBatch;

    v8f acc[2][2];
    const v8f vzero = {0.f,0.f,0.f,0.f,0.f,0.f,0.f,0.f};
    acc[0][0] = vzero; acc[0][1] = vzero; acc[1][0] = vzero; acc[1][1] = vzero;

    const int half = lane >> 4;
    const int l16  = lane & 15;

    for (int k0 = 0; k0 < K; k0 += TK) {
        // ---- stage A tile (TM x TK) as f16, float4 global reads ----
        if (ATRANS) {
            // A[k][m], m contiguous: 16 float4 per k-row
            for (int e = tid; e < (TM / 4) * TK; e += 256) {
                int k  = e >> 4;
                int m4 = (e & 15) * 4;
                int gk = k0 + k, gm = m0 + m4;
                float4 v = {0.f, 0.f, 0.f, 0.f};
                if (gk < K) {
                    if (gm + 3 < M) {
                        v = *(const float4*)&Ab[(size_t)gk * lda + gm];
                    } else {
                        const float* p = &Ab[(size_t)gk * lda];
                        if (gm + 0 < M) v.x = p[gm + 0];
                        if (gm + 1 < M) v.y = p[gm + 1];
                        if (gm + 2 < M) v.z = p[gm + 2];
                        if (gm + 3 < M) v.w = p[gm + 3];
                    }
                }
                As[(m4 + 0) * LDSK + k] = (_Float16)v.x;
                As[(m4 + 1) * LDSK + k] = (_Float16)v.y;
                As[(m4 + 2) * LDSK + k] = (_Float16)v.z;
                As[(m4 + 3) * LDSK + k] = (_Float16)v.w;
            }
        } else {
            // A[m][k], k contiguous: 8 float4 per m-row, contiguous LDS store
            for (int e = tid; e < TM * (TK / 4); e += 256) {
                int m  = e >> 3;
                int k4 = (e & 7) * 4;
                int gm = m0 + m, gk = k0 + k4;
                float4 v = {0.f, 0.f, 0.f, 0.f};
                if (gm < M) {
                    if (gk + 3 < K) {
                        v = *(const float4*)&Ab[(size_t)gm * lda + gk];
                    } else {
                        const float* p = &Ab[(size_t)gm * lda];
                        if (gk + 0 < K) v.x = p[gk + 0];
                        if (gk + 1 < K) v.y = p[gk + 1];
                        if (gk + 2 < K) v.z = p[gk + 2];
                        if (gk + 3 < K) v.w = p[gk + 3];
                    }
                }
                v4h h = { (_Float16)v.x, (_Float16)v.y, (_Float16)v.z, (_Float16)v.w };
                *(v4h*)&As[m * LDSK + k4] = h;   // 8B-aligned ds_store_b64
            }
        }
        // ---- stage B tile (store transposed [n][k]) ----
        if (CONV3) {
            for (int e = tid; e < TN * TK; e += 256) {
                int k = e >> 7, n = e & 127;
                int gn = n0 + n, gk = k0 + k;
                float v = 0.f;
                if (gn < N && gk < K) {
                    int ic = gk / 9, r = gk % 9;
                    int dy = r / 3 - 1, dx = r % 3 - 1;
                    int y = gn / Wd + dy, x = gn % Wd + dx;
                    if (y >= 0 && y < Hd && x >= 0 && x < Wd)
                        v = Bb[(size_t)ic * (Hd * Wd) + (size_t)y * Wd + x];
                }
                Bs[n * LDSK + k] = (_Float16)v;
            }
        } else {
            // B[k][n], n contiguous: 32 float4 per k-row
            for (int e = tid; e < TK * (TN / 4); e += 256) {
                int k  = e >> 5;
                int n4 = (e & 31) * 4;
                int gk = k0 + k, gn = n0 + n4;
                float4 v = {0.f, 0.f, 0.f, 0.f};
                if (gk < K) {
                    if (gn + 3 < N) {
                        v = *(const float4*)&Bb[(size_t)gk * N + gn];
                        if (gk + TK < K)  // speculative prefetch of next K tile
                            __builtin_prefetch(&Bb[(size_t)(gk + TK) * N + gn], 0, 1);
                    } else {
                        const float* p = &Bb[(size_t)gk * N];
                        if (gn + 0 < N) v.x = p[gn + 0];
                        if (gn + 1 < N) v.y = p[gn + 1];
                        if (gn + 2 < N) v.z = p[gn + 2];
                        if (gn + 3 < N) v.w = p[gn + 3];
                    }
                }
                Bs[(n4 + 0) * LDSK + k] = (_Float16)v.x;
                Bs[(n4 + 1) * LDSK + k] = (_Float16)v.y;
                Bs[(n4 + 2) * LDSK + k] = (_Float16)v.z;
                Bs[(n4 + 3) * LDSK + k] = (_Float16)v.w;
            }
        }
        __syncthreads();

        // ---- fragments + WMMA ----
        union { v8h h[2]; v16h v; } av[2], bv[2];
#pragma unroll
        for (int f = 0; f < 2; ++f) {
            int am = wm * 32 + f * 16 + l16;
            av[f].h[0] = *(const v8h*)&As[am * LDSK + half * 8];
            av[f].h[1] = *(const v8h*)&As[am * LDSK + 16 + half * 8];
        }
#pragma unroll
        for (int g = 0; g < 2; ++g) {
            int bn = wn * 32 + g * 16 + l16;
            bv[g].h[0] = *(const v8h*)&Bs[bn * LDSK + half * 8];
            bv[g].h[1] = *(const v8h*)&Bs[bn * LDSK + 16 + half * 8];
        }
#pragma unroll
        for (int f = 0; f < 2; ++f)
#pragma unroll
            for (int g = 0; g < 2; ++g)
                acc[f][g] = __builtin_amdgcn_wmma_f32_16x16x32_f16(
                    false, av[f].v, false, bv[g].v,
                    (short)0, acc[f][g], false, false);
        __syncthreads();
    }

    // ---- epilogue: C layout lane(0-15): M=r, lane(16-31): M=8+r, N = lane&15
#pragma unroll
    for (int f = 0; f < 2; ++f) {
#pragma unroll
        for (int g = 0; g < 2; ++g) {
#pragma unroll
            for (int r = 0; r < 8; ++r) {
                int m = m0 + wm * 32 + f * 16 + half * 8 + r;
                int n = n0 + wn * 32 + g * 16 + l16;
                if (m < M && n < N) {
                    float v = acc[f][g][r];
                    if (EPI == 1) {
                        v += bias[m];
                        v = v > 0.f ? v : 0.f;
                    } else if (EPI == 2) {
                        v = 0.5f * v + 0.5f;
                    }
                    Cb[(size_t)m * ldc + n] = v;
                }
            }
        }
    }
}

// ---------------------------------------------------------------------------
// Per-row top-4 threshold + min of local similarity (2304 f32 per row).
// The row is DMA'd into LDS via the Tensor Data Mover (1-D descriptor),
// then 8 waves reduce it with a sorted-4 LDS tree merge.
// ---------------------------------------------------------------------------
#define HWC 2304

__device__ __forceinline__ void top4_insert(float v, float& a0, float& a1, float& a2, float& a3) {
    if (v > a3) {
        if (v > a0)      { a3 = a2; a2 = a1; a1 = a0; a0 = v; }
        else if (v > a1) { a3 = a2; a2 = a1; a1 = v; }
        else if (v > a2) { a3 = a2; a2 = v; }
        else               a3 = v;
    }
}

__global__ __launch_bounds__(256)
void rowstats_k(const float* __restrict__ lsim, float* __restrict__ cutr,
                float* __restrict__ mnr, int HW)
{
    const int b = blockIdx.y, p = blockIdx.x, tid = threadIdx.x;
    const float* row = lsim + ((size_t)b * HW + p) * HW;

    float t0 = -3e38f, t1 = -3e38f, t2 = -3e38f, t3 = -3e38f, mn = 3e38f;

#ifdef ATHENA_TDM
    __shared__ __align__(16) float rowbuf[HWC];
    if (tid < 32) {                       // wave 0 issues one TDM op
        unsigned long long ga = (unsigned long long)(uintptr_t)row;
        unsigned ldsOff = (unsigned)(uintptr_t)&rowbuf[0];  // addr[31:0] = LDS byte offset
        // D# group 0: count=1, is_restore=0, gather off, type=2 ("image")
        u32x4 g0;
        g0[0] = 1u;
        g0[1] = ldsOff;
        g0[2] = (unsigned)(ga & 0xffffffffu);
        g0[3] = (unsigned)((ga >> 32) & 0x1ffffffu) | (2u << 30);
        // D# group 1: data_size=2 (4B), tensor_dim0=HW, tensor_dim1=1,
        // tile_dim0=HW, tile_dim1=0 (1-D), stride0=HW
        i32x8 g1;
        g1[0] = (2 << 16);
        g1[1] = (int)((unsigned)(HW & 0xffff) << 16);
        g1[2] = (int)(((unsigned)HW >> 16) | (1u << 16));
        g1[3] = (int)((unsigned)(HW & 0xffff) << 16);
        g1[4] = 0;
        g1[5] = HW;
        g1[6] = 0;
        g1[7] = 0;
        i32x4 gz4 = {0, 0, 0, 0};
        i32x8 gz8 = {0, 0, 0, 0, 0, 0, 0, 0};
        __builtin_amdgcn_tensor_load_to_lds(g0, g1, gz4, gz4, gz8, 0);
        __builtin_amdgcn_s_wait_tensorcnt(0);
    }
    __syncthreads();
    for (int q = tid; q < HW; q += 256) {
        float v = rowbuf[q];
        mn = fminf(mn, v);
        top4_insert(v, t0, t1, t2, t3);
    }
#else
    for (int q = tid; q < HW; q += 256) {
        float v = row[q];
        mn = fminf(mn, v);
        top4_insert(v, t0, t1, t2, t3);
    }
#endif

    __shared__ float st[256][4];
    __shared__ float sm[256];
    st[tid][0] = t0; st[tid][1] = t1; st[tid][2] = t2; st[tid][3] = t3; sm[tid] = mn;
    __syncthreads();
    for (int s = 128; s > 0; s >>= 1) {
        if (tid < s) {
            float a0 = st[tid][0], a1 = st[tid][1], a2 = st[tid][2], a3 = st[tid][3];
#pragma unroll
            for (int j = 0; j < 4; ++j)
                top4_insert(st[tid + s][j], a0, a1, a2, a3);
            st[tid][0] = a0; st[tid][1] = a1; st[tid][2] = a2; st[tid][3] = a3;
            sm[tid] = fminf(sm[tid], sm[tid + s]);
        }
        __syncthreads();
    }
    if (tid == 0) {
        cutr[(size_t)b * HW + p] = st[0][3];  // 4th largest of raw lsim row
        mnr [(size_t)b * HW + p] = sm[0];
    }
}

// ---------------------------------------------------------------------------
// Matching scores: for column q,
//   global:  max_p gsim[p][q] * seg[p]
//   local :  max_p seg[p] * (lsim[p][q] >= cut[p] ? lsim[p][q] : mn[p])
// (seg >= 0, so row scaling commutes with the top-4 threshold; seg==0 -> 0.)
// Writes 4 channels directly into x1 concat buffer at channel offset 256.
// Requires HW % 256 == 0 (2304 = 9*256).
// ---------------------------------------------------------------------------
__global__ __launch_bounds__(256)
void score_k(const float* __restrict__ gsim, const float* __restrict__ lsim,
             const float* __restrict__ cutr, const float* __restrict__ mnr,
             const float* __restrict__ iseg, const float* __restrict__ pseg,
             float* __restrict__ x1, int HW, long long x1Batch)
{
    const int b = blockIdx.y;
    const int q = blockIdx.x * 256 + threadIdx.x;
    const int tid = threadIdx.x;
    __shared__ float sg0[256], sg1[256], sl0[256], sl1[256], sc[256], sm_[256];
    float g0 = -3e38f, g1 = -3e38f, l0 = -3e38f, l1 = -3e38f;
    for (int p0 = 0; p0 < HW; p0 += 256) {
        int pl = p0 + tid;
        sg0[tid] = iseg[(size_t)b * 2 * HW + pl];
        sg1[tid] = iseg[(size_t)b * 2 * HW + HW + pl];
        sl0[tid] = pseg[(size_t)b * 2 * HW + pl];
        sl1[tid] = pseg[(size_t)b * 2 * HW + HW + pl];
        sc [tid] = cutr[(size_t)b * HW + pl];
        sm_[tid] = mnr [(size_t)b * HW + pl];
        __syncthreads();
        for (int j = 0; j < 256; ++j) {
            size_t base = ((size_t)b * HW + (size_t)(p0 + j)) * HW + q;
            float gv = gsim[base];
            float lv = lsim[base];
            g0 = fmaxf(g0, gv * sg0[j]);
            g1 = fmaxf(g1, gv * sg1[j]);
            float lt = (lv >= sc[j]) ? lv : sm_[j];
            l0 = fmaxf(l0, sl0[j] * lt);
            l1 = fmaxf(l1, sl1[j] * lt);
        }
        __syncthreads();
    }
    float* xb = x1 + (size_t)b * x1Batch + (size_t)256 * HW + q;
    xb[0]              = g0;
    xb[(size_t)HW]     = g1;
    xb[(size_t)2 * HW] = l0;
    xb[(size_t)3 * HW] = l1;
}

// ---------------------------------------------------------------------------
// Copy mask_feats (B,256,HW) into x1 channels [260, 516).
// ---------------------------------------------------------------------------
__global__ void maskcopy_k(const float* __restrict__ mf, float* __restrict__ x1,
                           int HW, long long x1Batch)
{
    int idx = blockIdx.x * 256 + threadIdx.x;
    int total = 2 * 256 * HW;
    if (idx >= total) return;
    int hw = idx % HW; int t = idx / HW;
    int c  = t % 256;  int b = t / 256;
    x1[(size_t)b * x1Batch + (size_t)(260 + c) * HW + hw] = mf[idx];
}

// ---------------------------------------------------------------------------
// ConvTranspose2d, outC = 2 (torch weight layout (inC, 2, k, k)).
// oy = iy*stride + ky - pad. Naive per-output-pixel; negligible FLOPs.
// ---------------------------------------------------------------------------
__global__ void deconv_k(const float* __restrict__ in, const float* __restrict__ w,
                         const float* __restrict__ bias, float* __restrict__ out,
                         int inC, int Hin, int Win, int Hout, int Wout,
                         int ksz, int stride, int pad,
                         long long inBatch, long long outBatch, int total)
{
    int idx = blockIdx.x * 256 + threadIdx.x;
    if (idx >= total) return;
    int ox = idx % Wout; int t = idx / Wout;
    int oy = t % Hout;   t /= Hout;
    int oc = t & 1;      int b = t >> 1;
    float accv = bias[oc];
    for (int ky = 0; ky < ksz; ++ky) {
        int ty = oy + pad - ky;
        if (ty < 0 || (ty % stride)) continue;
        int iy = ty / stride; if (iy >= Hin) continue;
        for (int kx = 0; kx < ksz; ++kx) {
            int tx = ox + pad - kx;
            if (tx < 0 || (tx % stride)) continue;
            int ix = tx / stride; if (ix >= Win) continue;
            const float* ip = in + (size_t)b * inBatch + (size_t)iy * Win + ix;
            const float* wp = w + ((size_t)oc * ksz + ky) * ksz + kx;
            float s = 0.f;
            for (int ic = 0; ic < inC; ++ic)
                s += ip[(size_t)ic * Hin * Win] * wp[(size_t)ic * 2 * ksz * ksz];
            accv += s;
        }
    }
    out[(size_t)b * outBatch + (size_t)oc * Hout * Wout + (size_t)oy * Wout + ox] = accv;
}

// ---------------------------------------------------------------------------
// Host launcher
// ---------------------------------------------------------------------------
static inline size_t alignup(size_t x) { return (x + 255) & ~(size_t)255; }

extern "C" void kernel_launch(void* const* d_in, const int* in_sizes, int n_in,
                              void* d_out, int out_size, void* d_ws, size_t ws_size,
                              hipStream_t stream)
{
    (void)in_sizes; (void)n_in; (void)out_size; (void)ws_size;

    const int Bn = 2, CK = 512, H16 = 48, HW = H16 * H16;           // 2304
    const int H8 = 96,  HW8 = H8 * H8;                              // 9216
    const int H4 = 192, HW4 = H4 * H4;                              // 36864
    const int HO = 768;

    const float* feats_s16 = (const float*)d_in[0];
    const float* feats_s8  = (const float*)d_in[1];
    const float* feats_s4  = (const float*)d_in[2];
    const float* key       = (const float*)d_in[3];
    const float* init_key  = (const float*)d_in[4];
    const float* prev_key  = (const float*)d_in[5];
    const float* init_seg  = (const float*)d_in[6];
    const float* prev_seg  = (const float*)d_in[7];
    const float* mask_f    = (const float*)d_in[8];
    const float* w_conv1 = (const float*)d_in[9],  *b_conv1 = (const float*)d_in[10];
    const float* w_blend1= (const float*)d_in[11], *b_blend1= (const float*)d_in[12];
    const float* w_dec1  = (const float*)d_in[13], *b_dec1  = (const float*)d_in[14];
    const float* w_conv2 = (const float*)d_in[15], *b_conv2 = (const float*)d_in[16];
    const float* w_blend2= (const float*)d_in[17], *b_blend2= (const float*)d_in[18];
    const float* w_dec2  = (const float*)d_in[19], *b_dec2  = (const float*)d_in[20];
    const float* w_conv3 = (const float*)d_in[21], *b_conv3 = (const float*)d_in[22];
    const float* w_blend3= (const float*)d_in[23], *b_blend3= (const float*)d_in[24];
    const float* w_dec3  = (const float*)d_in[25], *b_dec3  = (const float*)d_in[26];
    float* out = (float*)d_out;
    char*  ws  = (char*)d_ws;

    // ---- workspace layout (with buffer reuse; peak ~152 MB) ----
    const size_t F = sizeof(float);
    const size_t simB = (size_t)Bn * HW * HW * F;          // 42.47 MB each
    size_t o_gsim = 0;
    size_t o_lsim = o_gsim + simB;
    size_t o_cut  = o_lsim + simB;
    size_t o_mn   = o_cut + (size_t)Bn * HW * F;
    size_t o_x1   = alignup(o_mn + (size_t)Bn * HW * F);   // concat1: 516 ch
    size_t x1B    = (size_t)Bn * 516 * HW * F;
    size_t o_t1   = 0;                                     // reuse gsim (dead)
    size_t t1B    = (size_t)Bn * 256 * HW * F;
    size_t o_x2   = alignup(o_t1 + t1B);                   // concat2: 258 ch
    size_t o_t2   = alignup(o_x1 + x1B);                   // after x1 dies
    size_t o_x3   = 0;                                     // reuse t1/x2 (dead)
    size_t x3B    = (size_t)Bn * 258 * HW4 * F;
    size_t o_t3   = alignup(o_x3 + x3B);                   // after t2 dies

    float* gsim = (float*)(ws + o_gsim);
    float* lsim = (float*)(ws + o_lsim);
    float* cutr = (float*)(ws + o_cut);
    float* mnr  = (float*)(ws + o_mn);
    float* x1   = (float*)(ws + o_x1);
    float* t1   = (float*)(ws + o_t1);
    float* x2   = (float*)(ws + o_x2);
    float* t2   = (float*)(ws + o_t2);
    float* x3   = (float*)(ws + o_x3);
    float* t3   = (float*)(ws + o_t3);

    dim3 blk(256);

    // ---- Phase 1: similarity GEMMs (A = ref key, K x M; B = query key) ----
    {
        dim3 g((HW + TN - 1) / TN, (HW + TM - 1) / TM, Bn);
        gemm_wmma_f16<1, 0, 2><<<g, blk, 0, stream>>>(
            init_key, (long long)CK * HW, HW, key, (long long)CK * HW,
            gsim, (long long)HW * HW, HW, HW, HW, CK, nullptr, 0, 0);
        gemm_wmma_f16<1, 0, 2><<<g, blk, 0, stream>>>(
            prev_key, (long long)CK * HW, HW, key, (long long)CK * HW,
            lsim, (long long)HW * HW, HW, HW, HW, CK, nullptr, 0, 0);
    }
    rowstats_k<<<dim3(HW, Bn), blk, 0, stream>>>(lsim, cutr, mnr, HW);

    // conv1 (1x1, 1024->256) writes x1 channels [0,256)
    {
        dim3 g((HW + TN - 1) / TN, (256 + TM - 1) / TM, Bn);
        gemm_wmma_f16<0, 0, 1><<<g, blk, 0, stream>>>(
            w_conv1, 0, 1024, feats_s16, (long long)1024 * HW,
            x1, (long long)516 * HW, HW, 256, HW, 1024, b_conv1, 0, 0);
    }
    score_k<<<dim3(HW / 256, Bn), blk, 0, stream>>>(gsim, lsim, cutr, mnr,
                                                    init_seg, prev_seg,
                                                    x1, HW, (long long)516 * HW);
    maskcopy_k<<<dim3((2 * 256 * HW + 255) / 256), blk, 0, stream>>>(
        mask_f, x1, HW, (long long)516 * HW);

    // blend1 (3x3, 516->256) -> t1 ; deconv1 -> x2 channels [256,258)
    {
        dim3 g((HW + TN - 1) / TN, (256 + TM - 1) / TM, Bn);
        gemm_wmma_f16<0, 1, 1><<<g, blk, 0, stream>>>(
            w_blend1, 0, 516 * 9, x1, (long long)516 * HW,
            t1, (long long)256 * HW, HW, 256, HW, 516 * 9, b_blend1, H16, H16);
    }
    {
        int total = Bn * 2 * HW8;
        deconv_k<<<dim3((total + 255) / 256), blk, 0, stream>>>(
            t1, w_dec1, b_dec1, x2 + (size_t)256 * HW8,
            256, H16, H16, H8, H8, 4, 2, 1,
            (long long)256 * HW, (long long)258 * HW8, total);
    }
    // conv2 (1x1, 512->256) writes x2 channels [0,256)
    {
        dim3 g((HW8 + TN - 1) / TN, (256 + TM - 1) / TM, Bn);
        gemm_wmma_f16<0, 0, 1><<<g, blk, 0, stream>>>(
            w_conv2, 0, 512, feats_s8, (long long)512 * HW8,
            x2, (long long)258 * HW8, HW8, 256, HW8, 512, b_conv2, 0, 0);
    }
    // blend2 (3x3, 258->256) -> t2
    {
        dim3 g((HW8 + TN - 1) / TN, (256 + TM - 1) / TM, Bn);
        gemm_wmma_f16<0, 1, 1><<<g, blk, 0, stream>>>(
            w_blend2, 0, 258 * 9, x2, (long long)258 * HW8,
            t2, (long long)256 * HW8, HW8, 256, HW8, 258 * 9, b_blend2, H8, H8);
    }
    // conv3 (1x1, 256->256) writes x3 channels [0,256) (x2/t1 dead)
    {
        dim3 g((HW4 + TN - 1) / TN, (256 + TM - 1) / TM, Bn);
        gemm_wmma_f16<0, 0, 1><<<g, blk, 0, stream>>>(
            w_conv3, 0, 256, feats_s4, (long long)256 * HW4,
            x3, (long long)258 * HW4, HW4, 256, HW4, 256, b_conv3, 0, 0);
    }
    // deconv2 -> x3 channels [256,258)
    {
        int total = Bn * 2 * HW4;
        deconv_k<<<dim3((total + 255) / 256), blk, 0, stream>>>(
            t2, w_dec2, b_dec2, x3 + (size_t)256 * HW4,
            256, H8, H8, H4, H4, 4, 2, 1,
            (long long)256 * HW8, (long long)258 * HW4, total);
    }
    // blend3 (3x3, 258->256) -> t3 (t2 dead after deconv2)
    {
        dim3 g((HW4 + TN - 1) / TN, (256 + TM - 1) / TM, Bn);
        gemm_wmma_f16<0, 1, 1><<<g, blk, 0, stream>>>(
            w_blend3, 0, 258 * 9, x3, (long long)258 * HW4,
            t3, (long long)256 * HW4, HW4, 256, HW4, 258 * 9, b_blend3, H4, H4);
    }
    // deconv3 (k=6, s=4, p=1) -> final output (B,2,768,768)
    {
        int total = Bn * 2 * HO * HO;
        deconv_k<<<dim3((total + 255) / 256), blk, 0, stream>>>(
            t3, w_dec3, b_dec3, out,
            256, H4, H4, HO, HO, 6, 4, 1,
            (long long)256 * HW4, (long long)2 * HO * HO, total);
    }
}
